// SAModule_43997644980917
// MI455X (gfx1250) — matrix operand order repaired and checked by reading, not compiled, not run on hardware
//
#include <hip/hip_runtime.h>
#include <hip/hip_bf16.h>

#define N_PTS 32768
#define F_IN  64
#define M_CTR 8192
#define K_NBR 64
#define RAD2  (0.08f * 0.08f)
#define H1_   64
#define H2_   64
#define H3_   128
#define CAP   160   // candidate cap per centroid (Poisson mean ~70; P(>160) ~ 0)

typedef __attribute__((ext_vector_type(16))) _Float16 v16h;
typedef __attribute__((ext_vector_type(8)))  _Float16 h8;
typedef __attribute__((ext_vector_type(4)))  _Float16 h4;
typedef __attribute__((ext_vector_type(8)))  float    v8f;
typedef __attribute__((ext_vector_type(4)))  float    f4;

// ---------------------------------------------------------------------------
// Kernel 1: farthest point sampling. Single workgroup, 1024 threads (32 waves,
// wave32). Each thread owns 32 points (strided j = i*1024 + t) in registers,
// and its slice of the running min-distance array in registers. Per iteration:
// broadcast current point, update d2, argmax via shfl + LDS cross-wave reduce.
// ---------------------------------------------------------------------------
__global__ __launch_bounds__(1024) void fps_kernel(const float* __restrict__ pos,
                                                   int* __restrict__ idx_out) {
  __shared__ float bc[3];
  __shared__ float wbv[32];
  __shared__ int   wbi[32];
  __shared__ int   bci;
  const int t = threadIdx.x;

  float px[32], py[32], pz[32], d2[32];
#pragma unroll
  for (int i = 0; i < 32; ++i) {
    int j = i * 1024 + t;
    px[i] = pos[3 * j];
    py[i] = pos[3 * j + 1];
    pz[i] = pos[3 * j + 2];
    d2[i] = __builtin_inff();
  }

  int cur = 0;
  for (int it = 0; it < M_CTR; ++it) {
    if (t == (cur & 1023)) {
      int s = cur >> 10;
      float bx = 0.f, by = 0.f, bz = 0.f;
#pragma unroll
      for (int i = 0; i < 32; ++i)
        if (s == i) { bx = px[i]; by = py[i]; bz = pz[i]; }
      bc[0] = bx; bc[1] = by; bc[2] = bz;
    }
    if (t == 0) idx_out[it] = cur;
    __syncthreads();
    float bx = bc[0], by = bc[1], bz = bc[2];

    float best = -1.f;
    int bidx = 0;
#pragma unroll
    for (int i = 0; i < 32; ++i) {
      float dx = px[i] - bx, dy = py[i] - by, dz = pz[i] - bz;
      float nd = dx * dx + dy * dy + dz * dz;
      float mn = d2[i] < nd ? d2[i] : nd;
      d2[i] = mn;
      if (mn > best) { best = mn; bidx = i * 1024 + t; }
    }
    // wave32 argmax reduce (tie -> lowest index, matching jnp.argmax)
#pragma unroll
    for (int o = 16; o > 0; o >>= 1) {
      float ov = __shfl_xor(best, o);
      int   oi = __shfl_xor(bidx, o);
      if (ov > best || (ov == best && oi < bidx)) { best = ov; bidx = oi; }
    }
    if ((t & 31) == 0) { wbv[t >> 5] = best; wbi[t >> 5] = bidx; }
    __syncthreads();
    if (t < 32) {
      float v = wbv[t];
      int   i = wbi[t];
#pragma unroll
      for (int o = 16; o > 0; o >>= 1) {
        float ov = __shfl_xor(v, o);
        int   oi = __shfl_xor(i, o);
        if (ov > v || (ov == v && oi < i)) { v = ov; i = oi; }
      }
      if (t == 0) bci = i;
    }
    __syncthreads();
    cur = bci;
  }
}

// ---------------------------------------------------------------------------
// Kernel 2: pre-swizzle W1/W2/W3 (f32 row-major [fin, fout]) into CDNA5 WMMA
// B-fragment layout (f16). Fragment = 32 lanes x 16 halves, lane stride padded
// to 24 halves (48B = odd multiple of 16B -> conflict-free b128 LDS loads).
// Lane n (0-15): column cb*16+n, rows kb*32+0..15 ; lane n (16-31): same col,
// rows kb*32+16..31. Frag ids: W1 kb*4+cb (12), W2 12+kb*4+cb (8),
// W3 20+kb*8+cb (16). W1 rows 67..95 zero-padded.
// ---------------------------------------------------------------------------
__global__ void prep_weights(const float* __restrict__ W1,
                             const float* __restrict__ W2,
                             const float* __restrict__ W3,
                             _Float16* __restrict__ blob) {
  int f = blockIdx.x;
  int l = threadIdx.x;
  const float* W;
  int fin, cols, kb, cb;
  if (f < 12)      { W = W1; fin = F_IN + 3; cols = H1_; kb = f >> 2;        cb = f & 3; }
  else if (f < 20) { W = W2; fin = H1_;      cols = H2_; kb = (f - 12) >> 2; cb = (f - 12) & 3; }
  else             { W = W3; fin = H2_;      cols = H3_; kb = (f - 20) >> 3; cb = (f - 20) & 7; }
  int col   = cb * 16 + (l & 15);
  int rbase = kb * 32 + ((l >> 4) << 4);
  _Float16* dst = blob + f * 768 + l * 24;
  for (int i = 0; i < 16; ++i) {
    int r = rbase + i;
    float v = (r < fin) ? W[r * cols + col] : 0.f;
    dst[i] = (_Float16)v;
  }
}

// ---------------------------------------------------------------------------
// Kernel 3: ball query. One wave per centroid (4 waves / 128-thread block).
// Scan all N points, ballot-compact within-radius candidates to per-wave LDS,
// then select the K=64 nearest (iterative wave-parallel min with registers).
// Also emits q and batch[idx] output sections.
// ---------------------------------------------------------------------------
__global__ __launch_bounds__(128) void ballq_kernel(
    const float* __restrict__ pos, const int* __restrict__ batch,
    const int* __restrict__ idx, int* __restrict__ nbr,
    int* __restrict__ cntArr, float* __restrict__ qArr,
    float* __restrict__ outQ, int* __restrict__ outB) {
  __shared__ int   sIdx[4 * CAP];
  __shared__ float sD2[4 * CAP];
  const int w = threadIdx.x >> 5, l = threadIdx.x & 31;
  const int m = blockIdx.x * 4 + w;
  int*   cI = sIdx + w * CAP;
  float* cD = sD2  + w * CAP;

  int ctr = idx[m];
  float qx = pos[3 * ctr], qy = pos[3 * ctr + 1], qz = pos[3 * ctr + 2];
  if (l == 0) {
    qArr[3 * m] = qx; qArr[3 * m + 1] = qy; qArr[3 * m + 2] = qz;
    outQ[3 * m] = qx; outQ[3 * m + 1] = qy; outQ[3 * m + 2] = qz;
    outB[m] = batch[ctr];
  }

  int cnt = 0;
  for (int base = 0; base < N_PTS; base += 32) {
    int j = base + l;
    float dx = pos[3 * j] - qx, dy = pos[3 * j + 1] - qy, dz = pos[3 * j + 2] - qz;
    float d2 = dx * dx + dy * dy + dz * dz;
    bool ok = d2 <= RAD2;
    unsigned mask = (unsigned)__ballot(ok);
    int off = __popc(mask & ((1u << l) - 1u));
    int p = cnt + off;
    if (ok && p < CAP) { cI[p] = j; cD[p] = d2; }
    cnt += __popc(mask);
  }
  int realCnt = cnt < CAP ? cnt : CAP;

  // candidates -> registers (5 slots per lane), then K-nearest selection
  float cv[5];
#pragma unroll
  for (int c = 0; c < 5; ++c) {
    int s = c * 32 + l;
    cv[c] = (s < realCnt) ? cD[s] : __builtin_inff();
  }
  int sel = realCnt < K_NBR ? realCnt : K_NBR;
  for (int k = 0; k < sel; ++k) {
    float bv = __builtin_inff();
    int   bs = 0x7fffffff;
#pragma unroll
    for (int c = 0; c < 5; ++c) {
      int s = c * 32 + l;
      if (cv[c] < bv || (cv[c] == bv && s < bs)) { bv = cv[c]; bs = s; }
    }
#pragma unroll
    for (int o = 16; o > 0; o >>= 1) {
      float ov = __shfl_xor(bv, o);
      int   os = __shfl_xor(bs, o);
      if (ov < bv || (ov == bv && os < bs)) { bv = ov; bs = os; }
    }
    if ((bs & 31) == l) {
#pragma unroll
      for (int c = 0; c < 5; ++c)
        if ((bs >> 5) == c) cv[c] = __builtin_inff();
    }
    if (l == 0) nbr[m * K_NBR + k] = cI[bs];
  }
  if (l == 0) cntArr[m] = sel;
}

// ---------------------------------------------------------------------------
// Kernel 4: PointConv MLP + masked max-pool via V_WMMA_F32_16X16X32_F16.
// One wave per centroid; 4 waves/WG. Dynamic LDS:
//   [weights blob 27648 halves][per-wave: feat 64x104 | h1 64x72 | h2 64x72]
// Row strides (104, 72 halves) are odd multiples of 16B -> conflict-free b128.
// ---------------------------------------------------------------------------
__global__ __launch_bounds__(128) void mlp_kernel(
    const float* __restrict__ x, const float* __restrict__ pos,
    const int* __restrict__ nbr, const int* __restrict__ cntArr,
    const float* __restrict__ qArr, const _Float16* __restrict__ blob,
    const float* __restrict__ B1, const float* __restrict__ B2,
    const float* __restrict__ B3, float* __restrict__ out) {
  extern __shared__ _Float16 sm[];
  const int tid = threadIdx.x;
  const int w = tid >> 5, l = tid & 31;
  const int m = blockIdx.x * 4 + w;

  // cooperative weight-blob load: 27648 halves = 3456 x 16B
  {
    const f4* src = (const f4*)blob;
    f4* dst = (f4*)sm;
    for (int i = tid; i < 3456; i += 128) dst[i] = src[i];
  }
  _Float16* feat = sm + 27648 + w * 15872;
  _Float16* hA   = feat + 64 * 104;
  _Float16* hB   = hA + 64 * 72;

  const int cnt = cntArr[m];
  const float q0 = qArr[3 * m], q1 = qArr[3 * m + 1], q2 = qArr[3 * m + 2];
  __syncthreads();

  // build feature tile: rows l and l+32; row = [x_j (64) | rel (3) | 0-pad]
  for (int rr = 0; rr < 2; ++rr) {
    int k = l + rr * 32;
    _Float16* row = feat + k * 104;
    h4 z = {(_Float16)0, (_Float16)0, (_Float16)0, (_Float16)0};
    if (k < cnt) {
      int j = nbr[m * K_NBR + k];
      const f4* xj = (const f4*)(x + j * F_IN);
#pragma unroll
      for (int f = 0; f < 16; ++f) {
        f4 v = xj[f];
        h4 hv;
        hv.x = (_Float16)v.x; hv.y = (_Float16)v.y;
        hv.z = (_Float16)v.z; hv.w = (_Float16)v.w;
        *(h4*)(row + f * 4) = hv;
      }
#pragma unroll
      for (int f = 16; f < 26; ++f) *(h4*)(row + f * 4) = z;
      row[64] = (_Float16)(pos[3 * j]     - q0);
      row[65] = (_Float16)(pos[3 * j + 1] - q1);
      row[66] = (_Float16)(pos[3 * j + 2] - q2);
    } else {
#pragma unroll
      for (int f = 0; f < 26; ++f) *(h4*)(row + f * 4) = z;
    }
  }
  __syncthreads();

  // A fragment: lanes 0-15 row=l&15, K = {o..o+15}, o=16*(l>=16)
  auto loadA = [&](const _Float16* buf, int stride, int rb, int kb) -> v16h {
    const _Float16* p = buf + (rb * 16 + (l & 15)) * stride + kb * 32 + ((l >> 4) << 4);
    h8 lo = *(const h8*)p;
    h8 hi = *(const h8*)(p + 8);
    return __builtin_shufflevector(lo, hi, 0, 1, 2, 3, 4, 5, 6, 7,
                                   8, 9, 10, 11, 12, 13, 14, 15);
  };
  auto loadB = [&](int f) -> v16h {
    const _Float16* p = sm + f * 768 + l * 24;
    h8 lo = *(const h8*)p;
    h8 hi = *(const h8*)(p + 8);
    return __builtin_shufflevector(lo, hi, 0, 1, 2, 3, 4, 5, 6, 7,
                                   8, 9, 10, 11, 12, 13, 14, 15);
  };

  // ---- layer 1: feat [64x96(pad)] x W1 [96x64] -> h1 (bias+ReLU, f16) ----
  for (int rb = 0; rb < 4; ++rb)
    for (int cb = 0; cb < 4; ++cb) {
      v8f acc = {};
      for (int kb = 0; kb < 3; ++kb) {
        v16h a = loadA(feat, 104, rb, kb);
        v16h b = loadB(kb * 4 + cb);
        acc = __builtin_amdgcn_wmma_f32_16x16x32_f16(false, a, false, b,
                                                     (short)0, acc, false, false);
      }
      int   col  = cb * 16 + (l & 15);
      int   row0 = rb * 16 + ((l >> 4) << 3);
      float bias = B1[col];
#pragma unroll
      for (int r = 0; r < 8; ++r) {
        float v = acc[r] + bias;
        v = v > 0.f ? v : 0.f;
        hA[(row0 + r) * 72 + col] = (_Float16)v;
      }
    }
  __syncthreads();

  // ---- layer 2: h1 [64x64] x W2 [64x64] -> h2 ----
  for (int rb = 0; rb < 4; ++rb)
    for (int cb = 0; cb < 4; ++cb) {
      v8f acc = {};
      for (int kb = 0; kb < 2; ++kb) {
        v16h a = loadA(hA, 72, rb, kb);
        v16h b = loadB(12 + kb * 4 + cb);
        acc = __builtin_amdgcn_wmma_f32_16x16x32_f16(false, a, false, b,
                                                     (short)0, acc, false, false);
      }
      int   col  = cb * 16 + (l & 15);
      int   row0 = rb * 16 + ((l >> 4) << 3);
      float bias = B2[col];
#pragma unroll
      for (int r = 0; r < 8; ++r) {
        float v = acc[r] + bias;
        v = v > 0.f ? v : 0.f;
        hB[(row0 + r) * 72 + col] = (_Float16)v;
      }
    }
  __syncthreads();

  // ---- layer 3: h2 [64x64] x W3 [64x128], bias+ReLU, mask rows>=cnt, max ----
  for (int cb = 0; cb < 8; ++cb) {
    float bias = B3[cb * 16 + (l & 15)];
    float cmax = -__builtin_inff();
    for (int rb = 0; rb < 4; ++rb) {
      v8f acc = {};
      for (int kb = 0; kb < 2; ++kb) {
        v16h a = loadA(hB, 72, rb, kb);
        v16h b = loadB(20 + kb * 8 + cb);
        acc = __builtin_amdgcn_wmma_f32_16x16x32_f16(false, a, false, b,
                                                     (short)0, acc, false, false);
      }
      int rowHalf = (l >> 4) << 3;
#pragma unroll
      for (int r = 0; r < 8; ++r) {
        float v = acc[r] + bias;
        v = v > 0.f ? v : 0.f;
        int row = rb * 16 + rowHalf + r;
        v = (row < cnt) ? v : -__builtin_inff();
        cmax = cmax > v ? cmax : v;
      }
    }
    float om = __shfl_xor(cmax, 16);
    cmax = cmax > om ? cmax : om;
    if (l < 16) out[m * H3_ + cb * 16 + l] = cmax;
  }
}

// ---------------------------------------------------------------------------
// launch: FPS -> weight swizzle -> ball query -> WMMA MLP + maxpool
// d_out = [out M*128 f32 | q M*3 f32 | batch[idx] M i32]
// ws:    [idx M i32 | cnt M i32 | nbr M*64 i32 | q M*3 f32 | blob 27648 f16]
// ---------------------------------------------------------------------------
extern "C" void kernel_launch(void* const* d_in, const int* in_sizes, int n_in,
                              void* d_out, int out_size, void* d_ws, size_t ws_size,
                              hipStream_t stream) {
  const float* x     = (const float*)d_in[0];
  const float* pos   = (const float*)d_in[1];
  const int*   batch = (const int*)d_in[2];
  const float* W1    = (const float*)d_in[3];
  const float* b1    = (const float*)d_in[4];
  const float* W2    = (const float*)d_in[5];
  const float* b2    = (const float*)d_in[6];
  const float* W3    = (const float*)d_in[7];
  const float* b3    = (const float*)d_in[8];
  (void)in_sizes; (void)n_in; (void)out_size; (void)ws_size; (void)b2;

  char* ws = (char*)d_ws;
  int*      idx  = (int*)ws;                                   // M
  int*      cnt  = (int*)(ws + 32768);                         // M
  int*      nbr  = (int*)(ws + 65536);                         // M*64
  float*    qArr = (float*)(ws + 65536 + 2097152);             // M*3
  _Float16* blob = (_Float16*)(ws + 65536 + 2097152 + 98304);  // 27648 halves

  float* out  = (float*)d_out;
  float* outQ = out + M_CTR * H3_;
  int*   outB = (int*)(out + M_CTR * H3_ + M_CTR * 3);

  fps_kernel<<<1, 1024, 0, stream>>>(pos, idx);
  prep_weights<<<36, 32, 0, stream>>>(W1, W2, W3, blob);
  ballq_kernel<<<M_CTR / 4, 128, 0, stream>>>(pos, batch, idx, nbr, cnt, qArr, outQ, outB);

  size_t smem = (size_t)(27648 + 4 * 15872) * sizeof(_Float16); // 182,272 B (< 320 KB/WGP)
  mlp_kernel<<<M_CTR / 4, 128, smem, stream>>>(x, pos, nbr, cnt, qArr, blob,
                                               b1, b2, b3, out);
}
